// fp_Fingerprint_88364657148417
// MI455X (gfx1250) — compile-verified, matrix operand order checked by reading, not run on hardware
//
#include <hip/hip_runtime.h>
#include <math.h>

// AttentiveFP fingerprint layer on gfx1250 — fully fused, one workgroup per
// molecule, f16 WMMA (v_wmma_f32_16x16x32_f16) with f32 accumulation.
// B=512, L=128, M=6, ATOM=39, BOND=10, F=256.

#define LDIM 128
#define MNBR 6
#define ADIM 39
#define BDIM 10
#define CDIM 49
#define FDIM 256
#define KATOM 64   // padded K for the K=39 / K=49 GEMMs (multiple of 32)

typedef __attribute__((ext_vector_type(16))) _Float16 v16h;
typedef __attribute__((ext_vector_type(8)))  _Float16 v8h;
typedef __attribute__((ext_vector_type(8)))  float    v8f;

// LDS row strides in halfs: multiples of 8 (16B rows for ds_load_b128),
// odd multiples of 8 so 16-row column accesses are bank-conflict free.
#define SA_LD 72
#define SB_LD 12
#define XN_LD 72
#define F_LD  264

// f16 weight workspace layout (offsets in halfs), K-padded with zeros
#define OFF_WATOM 0          // [256][64]
#define OFF_WNBR  16384      // [256][64]
#define OFF_WATT  32768      // [256][256]
#define OFF_WIH   98304      // [768][256]
#define OFF_WHH   294912     // [768][256]
#define WS_HALFS  491520     // ~0.94 MB of d_ws

#define SMEM_BYTES 274816    // ~268 KB dynamic LDS (<= 320 KB/WGP)

__device__ __forceinline__ float leakyf(float x) { return x > 0.f ? x : 0.01f * x; }
__device__ __forceinline__ float sigmf(float x)  { return 1.f / (1.f + __expf(-x)); }
__device__ __forceinline__ float eluf(float x)   { return x > 0.f ? x : (__expf(x) - 1.f); }

// 16-bit A/B fragment per ISA layout: lane holds K = base+0..7 and base+16..23
// (base = (lane>=16)*8), i.e. two contiguous 16B chunks -> 2x b128 loads.
__device__ __forceinline__ v16h load_frag(const _Float16* p) {
  v8h lo = *(const v8h*)(p);
  v8h hi = *(const v8h*)(p + 16);
  v16h r;
#pragma unroll
  for (int i = 0; i < 8; ++i) { r[i] = lo[i]; r[8 + i] = hi[i]; }
  return r;
}

__device__ __forceinline__ v8f wmma16(v16h a, v16h b, v8f c) {
  return __builtin_amdgcn_wmma_f32_16x16x32_f16(false, a, false, b, (short)0, c,
                                                false, false);
}

#define V8F_ZERO {0.f, 0.f, 0.f, 0.f, 0.f, 0.f, 0.f, 0.f}

// ---- weight f32 -> f16 (+K padding) into d_ws, run once per launch ----
__global__ __launch_bounds__(256) void cvt_weights(
    const float* __restrict__ W_atom, const float* __restrict__ W_nbr,
    const float* __restrict__ W_att, const float* __restrict__ W_ih,
    const float* __restrict__ W_hh, _Float16* __restrict__ ws) {
  for (int i = blockIdx.x * 256 + threadIdx.x; i < WS_HALFS; i += gridDim.x * 256) {
    float v;
    if (i < OFF_WNBR)      { int n = i >> 6, k = i & 63; v = (k < ADIM) ? W_atom[n * ADIM + k] : 0.f; }
    else if (i < OFF_WATT) { int j = i - OFF_WNBR; int n = j >> 6, k = j & 63; v = (k < CDIM) ? W_nbr[n * CDIM + k] : 0.f; }
    else if (i < OFF_WIH)  v = W_att[i - OFF_WATT];
    else if (i < OFF_WHH)  v = W_ih[i - OFF_WIH];
    else                   v = W_hh[i - OFF_WHH];
    ws[i] = (_Float16)v;
  }
}

__global__ __launch_bounds__(256) void attentivefp_fused(
    const float* __restrict__ atom_list, const float* __restrict__ bond_list,
    const int* __restrict__ adeg, const int* __restrict__ bdeg,
    const float* __restrict__ b_atom, const float* __restrict__ b_nbr,
    const float* __restrict__ W_align, const float* __restrict__ b_align,
    const float* __restrict__ b_attend, const float* __restrict__ b_ih,
    const float* __restrict__ b_hh, const _Float16* __restrict__ wsH,
    float* __restrict__ out) {
  extern __shared__ _Float16 smemh[];
  const int b = blockIdx.x, tid = threadIdx.x;
  const int wave = tid >> 5, lane = tid & 31;
  const int mn = lane & 15;
  const int kbase = (lane >> 4) << 3;  // 0 or 8 halfs into the K block
  const int mB = (lane >> 4) << 3;     // D-tile row base (0 or 8)

  _Float16* sA   = smemh;            // 128*72  = 9216 h
  _Float16* sBnd = sA + 9216;        // 128*12  = 1536 h
  _Float16* sAF  = sBnd + 1536;      // 128*264 = 33792 h  atom_feature (f16)
  _Float16* sG   = sAF + 33792;      // 33792 h            g = sum_m w*NF
  _Float16* sCTX = sG + 33792;       // 33792 h            elu(context)
  _Float16* sXn  = sCTX + 33792;     // 48*72  = 3456 h
  _Float16* sNF  = sXn + 3456;       // 48*264 = 12672 h
  int*   sDgA  = (int*)(sNF + 12672);   // 768
  int*   sDgB  = sDgA + 768;            // 768
  float* sWal  = (float*)(sDgB + 768);  // 512
  float* sBias = sWal + 512;            // 2304: atom|nbr|att|ih(768)|hh(768)
  float* sScore = sBias + 2304;         // 48
  float* sWgt   = sScore + 48;          // 48
  float* sSumW  = sWgt + 48;            // 128

  const _Float16* WATOM = wsH + OFF_WATOM;
  const _Float16* WNBR  = wsH + OFF_WNBR;
  const _Float16* WATT  = wsH + OFF_WATT;
  const _Float16* WIH   = wsH + OFF_WIH;
  const _Float16* WHH   = wsH + OFF_WHH;

  // ---- Phase 0: stage molecule (f32 -> f16), degrees, small f32 params ----
  {
    const float* aG = atom_list + (size_t)b * LDIM * ADIM;
    for (int i = tid; i < LDIM * ADIM; i += 256)
      sA[(i / ADIM) * SA_LD + (i % ADIM)] = (_Float16)aG[i];
    for (int i = tid; i < LDIM * (SA_LD - ADIM); i += 256) {  // zero K pad
      int r = i / (SA_LD - ADIM), c = ADIM + i % (SA_LD - ADIM);
      sA[r * SA_LD + c] = (_Float16)0.f;
    }
    const float* bG = bond_list + (size_t)b * LDIM * BDIM;
    for (int i = tid; i < LDIM * BDIM; i += 256)
      sBnd[(i / BDIM) * SB_LD + (i % BDIM)] = (_Float16)bG[i];
    for (int i = tid; i < LDIM * MNBR; i += 256) {
      sDgA[i] = adeg[(size_t)b * LDIM * MNBR + i];
      sDgB[i] = bdeg[(size_t)b * LDIM * MNBR + i];
    }
    for (int i = tid; i < 512; i += 256) sWal[i] = W_align[i];
    if (tid < 256) {
      sBias[tid] = b_atom[tid]; sBias[256 + tid] = b_nbr[tid];
      sBias[512 + tid] = b_attend[tid];
    }
    for (int i = tid; i < 768; i += 256) {
      sBias[768 + i] = b_ih[i]; sBias[1536 + i] = b_hh[i];
    }
  }
  __syncthreads();

  // ---- Phase 1: AF = leaky(atom @ W_atom^T + b) for all 128 atoms ----
  for (int ct = wave; ct < 16; ct += 8) {
    const _Float16* pB = WATOM + (size_t)(ct * 16 + mn) * KATOM + kbase;
    const int n = ct * 16 + mn;
    const float bb = sBias[n];
    for (int rg = 0; rg < 4; ++rg) {   // 2 row-tiles per B fragment
      const int r0 = rg * 32;
      v8f a0 = V8F_ZERO, a1 = V8F_ZERO;
      const _Float16* pA0 = sA + (r0 + mn) * SA_LD + kbase;
#pragma unroll
      for (int k0 = 0; k0 < KATOM; k0 += 32) {
        v16h bf = load_frag(pB + k0);
        a0 = wmma16(load_frag(pA0 + k0), bf, a0);
        a1 = wmma16(load_frag(pA0 + 16 * SA_LD + k0), bf, a1);
      }
#pragma unroll
      for (int i = 0; i < 8; ++i) {
        sAF[(r0 + mB + i) * F_LD + n]      = (_Float16)leakyf(a0[i] + bb);
        sAF[(r0 + 16 + mB + i) * F_LD + n] = (_Float16)leakyf(a1[i] + bb);
      }
    }
  }
  __syncthreads();

  const float bAlign = b_align[0];

  // ---- Phase 2: neighbors in 16 subtiles of 8 atoms (48 nbr rows) ----
  for (int s = 0; s < 16; ++s) {
    const int atom0 = s * 8;
    // gather concat(atom_nbr, bond_nbr) -> sXn [48 x 64], zero K pad
    for (int i = tid; i < 48 * KATOM; i += 256) {
      const int rr = i >> 6, c = i & 63;
      const int a = atom0 + rr / 6, m = rr % 6;
      _Float16 v;
      if (c < ADIM)      v = sA[sDgA[a * 6 + m] * SA_LD + c];
      else if (c < CDIM) v = sBnd[sDgB[a * 6 + m] * SB_LD + (c - ADIM)];
      else               v = (_Float16)0.f;
      sXn[rr * XN_LD + c] = v;
    }
    __syncthreads();
    // NF = leaky(Xn @ W_nbr^T + b): 3 row-tiles x 16 col-tiles
    for (int t = wave; t < 48; t += 8) {
      const int rt = t >> 4, ct = t & 15;
      v8f acc = V8F_ZERO;
      const _Float16* pA = sXn + (rt * 16 + mn) * XN_LD + kbase;
      const _Float16* pB = WNBR + (size_t)(ct * 16 + mn) * KATOM + kbase;
#pragma unroll
      for (int k0 = 0; k0 < KATOM; k0 += 32)
        acc = wmma16(load_frag(pA + k0), load_frag(pB + k0), acc);
      const int n = ct * 16 + mn;
      const float bb = sBias[256 + n];
#pragma unroll
      for (int i = 0; i < 8; ++i)
        sNF[(rt * 16 + mB + i) * F_LD + n] = (_Float16)leakyf(acc[i] + bb);
    }
    __syncthreads();
    // align score (f32), leaky, softmax mask
    if (tid < 48) {
      const int rr = tid, a = atom0 + rr / 6, m = rr % 6;
      float sc = bAlign;
      for (int c = 0; c < FDIM; ++c)
        sc += (float)sAF[a * F_LD + c] * sWal[c] +
              (float)sNF[rr * F_LD + c] * sWal[256 + c];
      sc = leakyf(sc);
      if (sDgA[a * 6 + m] == LDIM - 1) sc += -9e8f;
      sScore[rr] = sc;
    }
    __syncthreads();
    // masked softmax over 6 neighbors of 8 atoms
    if (tid < 8) {
      const int base = tid * 6, a = atom0 + tid;
      float mx = -1e30f;
      for (int m = 0; m < 6; ++m) mx = fmaxf(mx, sScore[base + m]);
      float e[6], sum = 0.f;
      for (int m = 0; m < 6; ++m) { e[m] = __expf(sScore[base + m] - mx); sum += e[m]; }
      const float inv = 1.f / sum;
      float sw = 0.f;
      for (int m = 0; m < 6; ++m) {
        float w = e[m] * inv;
        if (sDgA[a * 6 + m] == LDIM - 1) w = 0.f;  // attend_mask
        sWgt[base + m] = w; sw += w;
      }
      sSumW[a] = sw;
    }
    __syncthreads();
    // g = sum_m w_m * NF_m  (defer linear W_attend to one GEMM)
    for (int i = tid; i < 8 * FDIM; i += 256) {
      const int al = i >> 8, c = i & 255;
      float g = 0.f;
      for (int m = 0; m < 6; ++m)
        g += sWgt[al * 6 + m] * (float)sNF[(al * 6 + m) * F_LD + c];
      sG[(atom0 + al) * F_LD + c] = (_Float16)g;
    }
    __syncthreads();
  }

  // ---- Phase 3: context = elu(g @ W_attend^T + sumw*b_attend) ----
  for (int ct = wave; ct < 16; ct += 8) {
    const _Float16* pB = WATT + (size_t)(ct * 16 + mn) * FDIM + kbase;
    const int n = ct * 16 + mn;
    const float ba = sBias[512 + n];
    for (int rg = 0; rg < 4; ++rg) {
      const int r0 = rg * 32;
      v8f a0 = V8F_ZERO, a1 = V8F_ZERO;
      const _Float16* pA0 = sG + (r0 + mn) * F_LD + kbase;
#pragma unroll
      for (int k0 = 0; k0 < FDIM; k0 += 32) {
        v16h bf = load_frag(pB + k0);
        a0 = wmma16(load_frag(pA0 + k0), bf, a0);
        a1 = wmma16(load_frag(pA0 + 16 * F_LD + k0), bf, a1);
      }
#pragma unroll
      for (int i = 0; i < 8; ++i) {
        const int m0 = r0 + mB + i, m1 = m0 + 16;
        sCTX[m0 * F_LD + n] = (_Float16)eluf(a0[i] + sSumW[m0] * ba);
        sCTX[m1 * F_LD + n] = (_Float16)eluf(a1[i] + sSumW[m1] * ba);
      }
    }
  }
  __syncthreads();

  // ---- Phase 4: GRU (x=context, h=AF), relu, store. 12 wmma per B fetch:
  // 6 gate panels (~48 KB f16 per col-tile, fits WGP$) x 2 row-tiles. ----
  for (int ct = wave; ct < 16; ct += 8) {
    const int n = ct * 16 + mn;  // output feature column
    const _Float16* pir = WIH + (size_t)(n)*FDIM + kbase;
    const _Float16* piz = WIH + (size_t)(256 + n) * FDIM + kbase;
    const _Float16* pin = WIH + (size_t)(512 + n) * FDIM + kbase;
    const _Float16* phr = WHH + (size_t)(n)*FDIM + kbase;
    const _Float16* phz = WHH + (size_t)(256 + n) * FDIM + kbase;
    const _Float16* phn = WHH + (size_t)(512 + n) * FDIM + kbase;
    const float bir = sBias[768 + n], biz = sBias[1024 + n], bin = sBias[1280 + n];
    const float bhr = sBias[1536 + n], bhz = sBias[1792 + n], bhn = sBias[2048 + n];
    for (int rg = 0; rg < 4; ++rg) {
      const int r0 = rg * 32;
      v8f ir0 = V8F_ZERO, ir1 = V8F_ZERO, iz0 = V8F_ZERO, iz1 = V8F_ZERO;
      v8f in0 = V8F_ZERO, in1 = V8F_ZERO, hr0 = V8F_ZERO, hr1 = V8F_ZERO;
      v8f hz0 = V8F_ZERO, hz1 = V8F_ZERO, hn0 = V8F_ZERO, hn1 = V8F_ZERO;
      const _Float16* pax0 = sCTX + (r0 + mn) * F_LD + kbase;
      const _Float16* pah0 = sAF + (r0 + mn) * F_LD + kbase;
#pragma unroll
      for (int k0 = 0; k0 < FDIM; k0 += 32) {
        v16h ax0 = load_frag(pax0 + k0);
        v16h ax1 = load_frag(pax0 + 16 * F_LD + k0);
        v16h ah0 = load_frag(pah0 + k0);
        v16h ah1 = load_frag(pah0 + 16 * F_LD + k0);
        v16h bf;
        bf = load_frag(pir + k0); ir0 = wmma16(ax0, bf, ir0); ir1 = wmma16(ax1, bf, ir1);
        bf = load_frag(piz + k0); iz0 = wmma16(ax0, bf, iz0); iz1 = wmma16(ax1, bf, iz1);
        bf = load_frag(pin + k0); in0 = wmma16(ax0, bf, in0); in1 = wmma16(ax1, bf, in1);
        bf = load_frag(phr + k0); hr0 = wmma16(ah0, bf, hr0); hr1 = wmma16(ah1, bf, hr1);
        bf = load_frag(phz + k0); hz0 = wmma16(ah0, bf, hz0); hz1 = wmma16(ah1, bf, hz1);
        bf = load_frag(phn + k0); hn0 = wmma16(ah0, bf, hn0); hn1 = wmma16(ah1, bf, hn1);
      }
#pragma unroll
      for (int i = 0; i < 8; ++i) {
        {
          const int m = r0 + mB + i;
          const float r = sigmf(ir0[i] + bir + hr0[i] + bhr);
          const float z = sigmf(iz0[i] + biz + hz0[i] + bhz);
          const float nn = tanhf(in0[i] + bin + r * (hn0[i] + bhn));
          const float hp = (float)sAF[m * F_LD + n];
          out[((size_t)b * LDIM + m) * FDIM + n] = fmaxf((1.f - z) * nn + z * hp, 0.f);
        }
        {
          const int m = r0 + 16 + mB + i;
          const float r = sigmf(ir1[i] + bir + hr1[i] + bhr);
          const float z = sigmf(iz1[i] + biz + hz1[i] + bhz);
          const float nn = tanhf(in1[i] + bin + r * (hn1[i] + bhn));
          const float hp = (float)sAF[m * F_LD + n];
          out[((size_t)b * LDIM + m) * FDIM + n] = fmaxf((1.f - z) * nn + z * hp, 0.f);
        }
      }
    }
  }
}

extern "C" void kernel_launch(void* const* d_in, const int* in_sizes, int n_in,
                              void* d_out, int out_size, void* d_ws, size_t ws_size,
                              hipStream_t stream) {
  (void)in_sizes; (void)n_in; (void)out_size; (void)ws_size;
  const float* atom_list = (const float*)d_in[0];
  const float* bond_list = (const float*)d_in[1];
  const int*   adeg      = (const int*)d_in[2];
  const int*   bdeg      = (const int*)d_in[3];
  // d_in[4] atom_mask: unused by the reference computation
  const float* W_atom   = (const float*)d_in[5];
  const float* b_atom   = (const float*)d_in[6];
  const float* W_nbr    = (const float*)d_in[7];
  const float* b_nbr    = (const float*)d_in[8];
  const float* W_align  = (const float*)d_in[9];
  const float* b_align  = (const float*)d_in[10];
  const float* W_attend = (const float*)d_in[11];
  const float* b_attend = (const float*)d_in[12];
  const float* W_ih     = (const float*)d_in[13];
  const float* b_ih     = (const float*)d_in[14];
  const float* W_hh     = (const float*)d_in[15];
  const float* b_hh     = (const float*)d_in[16];
  float* out = (float*)d_out;
  _Float16* wsH = (_Float16*)d_ws;  // needs ~0.94 MB of scratch

  cvt_weights<<<dim3(512), dim3(256), 0, stream>>>(W_atom, W_nbr, W_attend, W_ih,
                                                   W_hh, wsH);
  attentivefp_fused<<<dim3(512), dim3(256), SMEM_BYTES, stream>>>(
      atom_list, bond_list, adeg, bdeg, b_atom, b_nbr, W_align, b_align,
      b_attend, b_ih, b_hh, wsH, out);
}
